// DGT_concat_sim_61211873903546
// MI455X (gfx1250) — matrix operand order repaired and checked by reading, not compiled, not run on hardware
//
#include <hip/hip_runtime.h>
#include <hip/hip_bf16.h>
#include <math.h>

// ---------------------------------------------------------------------------
// Problem constants (from the reference)
// ---------------------------------------------------------------------------
#define BN_   1536          // 32 graphs * 48 nodes
#define E_    72192         // 32 * 48 * 47 edges
#define NPG_  48            // nodes per graph
#define EPG_  2256          // edges per graph (48*47)
#define ND_   256
#define ED_   64
#define TD_   1024
#define HEADS_ 8
#define HD_    32
#define CAT_LD 96           // padded lda for [dist|edge_attr] (K=65)
#define HIN_LD 608          // padded lda for h_input          (K=577)

typedef __attribute__((ext_vector_type(16))) _Float16 v16h;
typedef __attribute__((ext_vector_type(4)))  _Float16 v4h;
typedef __attribute__((ext_vector_type(8)))  float    v8f;
typedef __attribute__((ext_vector_type(4)))  float    v4f;

__device__ __forceinline__ float silu_f(float x) { return x / (1.f + __expf(-x)); }

// A-fragment LDS position: element k (0..31) of a row -> position k, so
// lane-half h reads k = h*16 .. h*16+15 contiguously (same order as B).
__device__ __forceinline__ int a_pos(int k) {
  return k;
}

// ---------------------------------------------------------------------------
// WMMA GEMM:  C[MxN] = act_out( act_in(A)[MxK] @ W[KxN] + bias[N] )
// Requirements (all call sites satisfy): M % 128 == 0, N % 64 == 0.
// K arbitrary; lda = row stride of A (pad A with zeros if lda > K).
// Block = 256 threads = 8 waves; block tile 128x64; wave tile 32x32
// (4 chained V_WMMA_F32_16X16X32_F16 per 32-wide K slab, 2x A / 2x B reuse).
// LDS double-buffered and stored in fragment order: each lane reads its A/B
// fragments as contiguous 32-byte vectors (ds_load_b128 pairs); next slab is
// staged while current slab's WMMAs execute (one barrier per slab).
// ---------------------------------------------------------------------------
template <int ACT_IN, int ACT_OUT>
__global__ __launch_bounds__(256) void wmma_gemm(
    const float* __restrict__ A, const float* __restrict__ W,
    const float* __restrict__ bias, float* __restrict__ C,
    int M, int N, int K, int lda)
{
  __shared__ _Float16 sA[2][128][40];  // [buf][m][k]
  __shared__ _Float16 sB[2][64][40];   // [buf][n][k] (transposed)

  const int tid  = threadIdx.x;
  const int lane = tid & 31;
  const int wave = tid >> 5;
  const int wr   = wave >> 1;          // 0..3  wave row (32 rows each)
  const int wc   = wave & 1;           // 0..1  wave col (32 cols each)
  const int half = lane >> 4;          // 0/1
  const int l16  = lane & 15;

  const int m0 = blockIdx.y * 128;
  const int n0 = blockIdx.x * 64;

  const bool a_vec = ((lda & 3) == 0);

  // stage one 32-wide K slab (A: 128x32, B: 32x64) into LDS buffer `buf`
  auto stage = [&](int buf, int kk) {
    const bool full = (kk + 32 <= K);
    if (full && a_vec) {
      #pragma unroll
      for (int i = 0; i < 4; ++i) {
        int idx = tid + i * 256;                  // 0..1023 float4 chunks
        int am = idx >> 3, ak4 = (idx & 7) * 4;
        const float* ap = A + (size_t)(m0 + am) * lda + (kk + ak4);
        if (kk + 64 < K) __builtin_prefetch(ap + 32, 0, 3);   // slab after next
        v4f f = *(const v4f*)ap;
        if (ACT_IN) { f.x = silu_f(f.x); f.y = silu_f(f.y); f.z = silu_f(f.z); f.w = silu_f(f.w); }
        v4h h4; h4[0] = (_Float16)f.x; h4[1] = (_Float16)f.y;
        h4[2] = (_Float16)f.z; h4[3] = (_Float16)f.w;
        *(v4h*)&sA[buf][am][a_pos(ak4)] = h4;     // 4 contiguous frag slots
      }
    } else {
      #pragma unroll
      for (int i = 0; i < 16; ++i) {
        int idx = tid + i * 256;                  // 0..4095 scalars
        int am = idx >> 5, ak = idx & 31;
        int gk = kk + ak;
        float v = 0.f;
        if (gk < K) {
          v = A[(size_t)(m0 + am) * lda + gk];
          if (ACT_IN) v = silu_f(v);
        }
        sA[buf][am][a_pos(ak)] = (_Float16)v;
      }
    }
    if (full) {
      #pragma unroll
      for (int i = 0; i < 2; ++i) {
        int n = tid & 63;
        int k4 = ((tid >> 6) << 2) + i * 16;      // 0,4,8,12 (+16)
        const float* wp = W + (size_t)(kk + k4) * N + (n0 + n);
        if (kk + 64 < K) __builtin_prefetch(wp + (size_t)32 * N, 0, 3);
        v4h h4;
        h4[0] = (_Float16)wp[0];
        h4[1] = (_Float16)wp[(size_t)N];
        h4[2] = (_Float16)wp[(size_t)2 * N];
        h4[3] = (_Float16)wp[(size_t)3 * N];
        *(v4h*)&sB[buf][n][k4] = h4;
      }
    } else {
      #pragma unroll
      for (int i = 0; i < 8; ++i) {
        int idx = tid + i * 256;
        int bk = idx >> 6, bn = idx & 63;
        int gk = kk + bk;
        float v = (gk < K) ? W[(size_t)gk * N + (n0 + bn)] : 0.f;
        sB[buf][bn][bk] = (_Float16)v;
      }
    }
  };

  v8f acc00 = {}, acc01 = {}, acc10 = {}, acc11 = {};

  stage(0, 0);
  int p = 0;
  for (int k0 = 0; k0 < K; k0 += 32) {
    __syncthreads();                    // staged slab `p` now visible
    const int am0 = wr * 32 + l16;
    const int bn0 = wc * 32 + l16;
    v16h af0 = *(const v16h*)&sA[p][am0][half * 16];
    v16h af1 = *(const v16h*)&sA[p][am0 + 16][half * 16];
    v16h bf0 = *(const v16h*)&sB[p][bn0][half * 16];
    v16h bf1 = *(const v16h*)&sB[p][bn0 + 16][half * 16];
    if (k0 + 32 < K) stage(p ^ 1, k0 + 32);   // overlap staging with WMMA
    acc00 = __builtin_amdgcn_wmma_f32_16x16x32_f16(false, af0, false, bf0,
                                                   (short)0, acc00, false, false);
    acc01 = __builtin_amdgcn_wmma_f32_16x16x32_f16(false, af0, false, bf1,
                                                   (short)0, acc01, false, false);
    acc10 = __builtin_amdgcn_wmma_f32_16x16x32_f16(false, af1, false, bf0,
                                                   (short)0, acc10, false, false);
    acc11 = __builtin_amdgcn_wmma_f32_16x16x32_f16(false, af1, false, bf1,
                                                   (short)0, acc11, false, false);
    p ^= 1;
  }

  // ---- epilogue: C/D layout: VGPR r -> M = r + (lane>=16 ? 8 : 0) --------
  const int cn0 = n0 + wc * 32 + l16;
  const float bv0 = bias ? bias[cn0] : 0.f;
  const float bv1 = bias ? bias[cn0 + 16] : 0.f;
  #pragma unroll
  for (int r = 0; r < 8; ++r) {
    int cm0 = m0 + wr * 32 + r + half * 8;
    float v00 = acc00[r] + bv0, v01 = acc01[r] + bv1;
    float v10 = acc10[r] + bv0, v11 = acc11[r] + bv1;
    if (ACT_OUT) { v00 = silu_f(v00); v01 = silu_f(v01); v10 = silu_f(v10); v11 = silu_f(v11); }
    C[(size_t)cm0 * N + cn0]             = v00;
    C[(size_t)cm0 * N + cn0 + 16]        = v01;
    C[(size_t)(cm0 + 16) * N + cn0]      = v10;
    C[(size_t)(cm0 + 16) * N + cn0 + 16] = v11;
  }
}

// ---------------------------------------------------------------------------
// Edge geometry: dist/cdiff + concat([dist, edge_attr]) with padded stride 96
// ---------------------------------------------------------------------------
__global__ __launch_bounds__(256) void k_edge_geom(
    const float* __restrict__ pos, const int* __restrict__ ei,
    const float* __restrict__ edge_attr, const float* __restrict__ coors_scale,
    float* __restrict__ cat_ea, float* __restrict__ dist, float* __restrict__ cdiff)
{
  int e = blockIdx.x * blockDim.x + threadIdx.x;
  if (e >= E_) return;
  int row = ei[e], col = ei[E_ + e];
  float dx = pos[row * 3 + 0] - pos[col * 3 + 0];
  float dy = pos[row * 3 + 1] - pos[col * 3 + 1];
  float dz = pos[row * 3 + 2] - pos[col * 3 + 2];
  float d2 = dx * dx + dy * dy + dz * dz;
  float ds = sqrtf(d2 + 1e-12f);
  dist[e] = ds;
  float nrm = sqrtf(d2);
  float inv = coors_scale[0] / fmaxf(nrm, 1e-8f);
  cdiff[e * 3 + 0] = dx * inv;
  cdiff[e * 3 + 1] = dy * inv;
  cdiff[e * 3 + 2] = dz * inv;
  size_t cb = (size_t)e * CAT_LD;
  cat_ea[cb] = ds;
  #pragma unroll 8
  for (int c = 0; c < ED_; ++c) cat_ea[cb + 1 + c] = edge_attr[(size_t)e * ED_ + c];
  #pragma unroll
  for (int c = 65; c < CAT_LD; ++c) cat_ea[cb + c] = 0.f;
}

// ---------------------------------------------------------------------------
// y = modulate(LayerNorm(x), shift, scale) [* mask]; one row per block,
// blockDim == D (power of two), dynamic smem = D floats.  In-place safe.
// ---------------------------------------------------------------------------
__global__ __launch_bounds__(256) void k_ln_mod(
    const float* __restrict__ x, const float* __restrict__ mod, int mod_stride,
    int shift_off, int scale_off, const float* __restrict__ mask,
    float* __restrict__ y, int D)
{
  extern __shared__ float red[];
  int r = blockIdx.x, t = threadIdx.x;
  size_t base = (size_t)r * D;
  float xv = x[base + t];
  red[t] = xv; __syncthreads();
  for (int s = D >> 1; s > 0; s >>= 1) { if (t < s) red[t] += red[t + s]; __syncthreads(); }
  float mean = red[0] / D; __syncthreads();
  float d = xv - mean;
  red[t] = d * d; __syncthreads();
  for (int s = D >> 1; s > 0; s >>= 1) { if (t < s) red[t] += red[t + s]; __syncthreads(); }
  float nx = d * rsqrtf(red[0] / D + 1e-6f);
  size_t mb = (size_t)r * mod_stride;
  float out = nx * (1.f + mod[mb + scale_off + t]) + mod[mb + shift_off + t];
  if (mask) out *= mask[r];
  y[base + t] = out;
}

// y = modulate(LayerNorm(base + gate*delta), shift, scale) [* mask]
__global__ __launch_bounds__(256) void k_gate_ln_mod(
    const float* __restrict__ base_, const float* __restrict__ delta,
    const float* __restrict__ mod, int mod_stride,
    int gate_off, int shift_off, int scale_off,
    const float* __restrict__ mask, float* __restrict__ y, int D)
{
  extern __shared__ float red[];
  int r = blockIdx.x, t = threadIdx.x;
  size_t base = (size_t)r * D;
  size_t mb = (size_t)r * mod_stride;
  float xv = base_[base + t] + mod[mb + gate_off + t] * delta[base + t];
  red[t] = xv; __syncthreads();
  for (int s = D >> 1; s > 0; s >>= 1) { if (t < s) red[t] += red[t + s]; __syncthreads(); }
  float mean = red[0] / D; __syncthreads();
  float d = xv - mean;
  red[t] = d * d; __syncthreads();
  for (int s = D >> 1; s > 0; s >>= 1) { if (t < s) red[t] += red[t + s]; __syncthreads(); }
  float nx = d * rsqrtf(red[0] / D + 1e-6f);
  float out = nx * (1.f + mod[mb + scale_off + t]) + mod[mb + shift_off + t];
  if (mask) out *= mask[r];
  y[base + t] = out;
}

// alpha[e,h] = dot32(q[col,h,:], k[row,h,:] * e0[e,h,:]) / sqrt(32)
__global__ __launch_bounds__(256) void k_alpha(
    const float* __restrict__ q, const float* __restrict__ kmat,
    const float* __restrict__ e0, const int* __restrict__ ei,
    float* __restrict__ alpha)
{
  int idx = blockIdx.x * blockDim.x + threadIdx.x;
  if (idx >= E_ * HEADS_) return;
  int e = idx >> 3, h = idx & 7;
  int row = ei[e], col = ei[E_ + e];
  const float* qp = q    + (size_t)col * ND_ + h * HD_;
  const float* kp = kmat + (size_t)row * ND_ + h * HD_;
  const float* ep = e0   + (size_t)e   * ND_ + h * HD_;
  float s = 0.f;
  #pragma unroll 8
  for (int d = 0; d < HD_; ++d) s += qp[d] * kp[d] * ep[d];
  alpha[idx] = s * 0.17677669529663687f;   // 1/sqrt(32)
}

// Per-target-node softmax over its 47 incoming edges + weighted aggregation.
// Edge ids are analytic: e = b*2256 + src*47 + (dst - (dst>src)).  No atomics.
__global__ __launch_bounds__(256) void k_attn_node(
    const float* __restrict__ alpha, const float* __restrict__ vmat,
    const float* __restrict__ e1, float* __restrict__ h_attn)
{
  int node = blockIdx.x;
  int t = threadIdx.x;
  int h = t >> 5;
  int b = node / NPG_, il = node % NPG_;
  size_t ebase = (size_t)b * EPG_;

  float mx = -1e30f;
  for (int s = 0; s < NPG_; ++s) {
    if (s == il) continue;
    size_t e = ebase + (size_t)s * 47 + (il - (il > s ? 1 : 0));
    mx = fmaxf(mx, alpha[e * HEADS_ + h]);
  }
  float den = 0.f;
  for (int s = 0; s < NPG_; ++s) {
    if (s == il) continue;
    size_t e = ebase + (size_t)s * 47 + (il - (il > s ? 1 : 0));
    den += __expf(alpha[e * HEADS_ + h] - mx);
  }
  float acc = 0.f;
  for (int s = 0; s < NPG_; ++s) {
    if (s == il) continue;
    size_t e = ebase + (size_t)s * 47 + (il - (il > s ? 1 : 0));
    int row = b * NPG_ + s;
    float att = __expf(alpha[e * HEADS_ + h] - mx) / (den + 1e-16f);
    acc += att * vmat[(size_t)row * ND_ + t] * e1[e * ND_ + t];
  }
  h_attn[(size_t)node * ND_ + t] = acc;
}

// hsum[e,c] = h_attn[row,c] + h_attn[col,c]
__global__ __launch_bounds__(256) void k_pair_sum(
    const float* __restrict__ h_attn, const int* __restrict__ ei,
    float* __restrict__ hsum)
{
  int idx = blockIdx.x * blockDim.x + threadIdx.x;
  if (idx >= E_ * ND_) return;
  int e = idx >> 8, c = idx & 255;
  int row = ei[e], col = ei[E_ + e];
  hsum[idx] = h_attn[(size_t)row * ND_ + c] + h_attn[(size_t)col * ND_ + c];
}

// h_out = (h_node2 + g_mlp * ff2) * mask
__global__ __launch_bounds__(256) void k_node_out(
    const float* __restrict__ h_node2, const float* __restrict__ ff2,
    const float* __restrict__ nt, const float* __restrict__ mask,
    float* __restrict__ out)
{
  int idx = blockIdx.x * blockDim.x + threadIdx.x;
  if (idx >= BN_ * ND_) return;
  int r = idx >> 8, c = idx & 255;
  out[idx] = (h_node2[idx] + nt[(size_t)r * (6 * ND_) + 5 * ND_ + c] * ff2[idx]) * mask[r];
}

// h_edge_out = h_e + e_g_mlp * ffe2
__global__ __launch_bounds__(256) void k_edge_out(
    const float* __restrict__ h_e, const float* __restrict__ ffe2,
    const float* __restrict__ et, float* __restrict__ out)
{
  int idx = blockIdx.x * blockDim.x + threadIdx.x;
  if (idx >= E_ * ED_) return;
  int e = idx >> 6, c = idx & 63;
  out[idx] = h_e[idx] + et[(size_t)e * (6 * ED_) + 5 * ED_ + c] * ffe2[idx];
}

// h_input[e, 0:577] = [h_out[row], h_out[col], h_edge_out[e], dist[e]]; pad->0
__global__ __launch_bounds__(256) void k_concat_hinput(
    const float* __restrict__ h_out, const float* __restrict__ h_edge_out,
    const float* __restrict__ dist, const int* __restrict__ ei,
    float* __restrict__ h_input)
{
  int idx = blockIdx.x * blockDim.x + threadIdx.x;
  if (idx >= E_ * HIN_LD) return;
  int e = idx / HIN_LD, c = idx - e * HIN_LD;
  float v;
  if (c < ND_) {
    v = h_out[(size_t)ei[e] * ND_ + c];
  } else if (c < 2 * ND_) {
    v = h_out[(size_t)ei[E_ + e] * ND_ + (c - ND_)];
  } else if (c < 2 * ND_ + ED_) {
    v = h_edge_out[(size_t)e * ED_ + (c - 2 * ND_)];
  } else if (c == 576) {
    v = dist[e];
  } else {
    v = 0.f;                                    // zero pad (577..607)
  }
  h_input[idx] = v;
}

__global__ __launch_bounds__(256) void k_copy(
    const float* __restrict__ src, float* __restrict__ dst, int n)
{
  int i = blockIdx.x * blockDim.x + threadIdx.x;
  if (i < n) dst[i] = src[i];
}

// inv = tanh(dot256(uc1[e], W_uc2)); pos_out[row] += cdiff[e] * inv
__global__ __launch_bounds__(256) void k_inv_scatter(
    const float* __restrict__ uc1, const float* __restrict__ Wuc2,
    const float* __restrict__ cdiff, const int* __restrict__ ei,
    float* __restrict__ pos_out)
{
  int e = blockIdx.x * blockDim.x + threadIdx.x;
  if (e >= E_) return;
  const float* up = uc1 + (size_t)e * ND_;
  float s = 0.f;
  #pragma unroll 8
  for (int i = 0; i < ND_; ++i) s += up[i] * Wuc2[i];
  float inv = tanhf(s);
  int row = ei[e];
  atomicAdd(&pos_out[row * 3 + 0], cdiff[e * 3 + 0] * inv);
  atomicAdd(&pos_out[row * 3 + 1], cdiff[e * 3 + 1] * inv);
  atomicAdd(&pos_out[row * 3 + 2], cdiff[e * 3 + 2] * inv);
}

// ---------------------------------------------------------------------------
// Host orchestration
// ---------------------------------------------------------------------------
#define GEMM(AI, AO, A, W, B, C, M, N, K, LDA)                                 \
  do {                                                                         \
    dim3 g(((N) + 63) / 64, ((M) + 127) / 128);                                \
    wmma_gemm<AI, AO><<<g, 256, 0, stream>>>((A), (W), (B), (C), (M), (N), (K), (LDA)); \
  } while (0)

extern "C" void kernel_launch(void* const* d_in, const int* in_sizes, int n_in,
                              void* d_out, int out_size, void* d_ws, size_t ws_size,
                              hipStream_t stream) {
  const float* pos       = (const float*)d_in[0];
  const float* h         = (const float*)d_in[1];
  const float* edge_attr = (const float*)d_in[2];
  const float* node_mask = (const float*)d_in[3];
  const float* node_time = (const float*)d_in[4];
  const float* edge_time = (const float*)d_in[5];
  const int*   ei        = (const int*)d_in[6];
  const float* W_ee = (const float*)d_in[7];  const float* b_ee = (const float*)d_in[8];
  const float* W_nt = (const float*)d_in[9];  const float* b_nt = (const float*)d_in[10];
  const float* W_et = (const float*)d_in[11]; const float* b_et = (const float*)d_in[12];
  const float* Wq   = (const float*)d_in[13]; const float* bq   = (const float*)d_in[14];
  const float* Wk   = (const float*)d_in[15]; const float* bk   = (const float*)d_in[16];
  const float* Wv   = (const float*)d_in[17]; const float* bv   = (const float*)d_in[18];
  const float* We0  = (const float*)d_in[19];
  const float* We1  = (const float*)d_in[20];
  const float* W_n2e = (const float*)d_in[21]; const float* b_n2e = (const float*)d_in[22];
  const float* W_ff1 = (const float*)d_in[23]; const float* b_ff1 = (const float*)d_in[24];
  const float* W_ff2 = (const float*)d_in[25]; const float* b_ff2 = (const float*)d_in[26];
  const float* W_ff3 = (const float*)d_in[27]; const float* b_ff3 = (const float*)d_in[28];
  const float* W_ff4 = (const float*)d_in[29]; const float* b_ff4 = (const float*)d_in[30];
  const float* coors_scale = (const float*)d_in[31];
  const float* W_ut = (const float*)d_in[32]; const float* b_ut = (const float*)d_in[33];
  const float* W_ui = (const float*)d_in[34]; const float* b_ui = (const float*)d_in[35];
  const float* W_uc1 = (const float*)d_in[36]; const float* b_uc1 = (const float*)d_in[37];
  const float* W_uc2 = (const float*)d_in[38];

  // outputs: h_out [BN,256] | h_edge_out [E,64] | pos_out [BN,3]
  float* out_h    = (float*)d_out;
  float* out_edge = out_h + (size_t)BN_ * ND_;
  float* out_pos  = out_edge + (size_t)E_ * ED_;

  // -------- workspace bump allocator (floats), with aliasing ---------------
  float* ws = (float*)d_ws;
  size_t o = 0;
  auto alloc = [&](size_t n) { float* p = ws + o; o += n; return p; };
  float* f_cat   = alloc((size_t)E_ * CAT_LD);   // dead after ea GEMM; reused: alpha
  float* f_dist  = alloc(E_);
  float* f_cdiff = alloc((size_t)E_ * 3);
  float* f_nt    = alloc((size_t)BN_ * 6 * ND_); // 1536 x 1536
  float* f_et    = alloc((size_t)E_ * 6 * ED_);  // E x 384; reused: ui (E x 256)
  float* f_ut    = alloc((size_t)E_ * 2 * ND_);  // E x 512
  float* f_ea    = alloc((size_t)E_ * ED_);
  float* f_hh    = alloc((size_t)BN_ * ND_);
  float* f_q     = alloc((size_t)BN_ * ND_);
  float* f_k     = alloc((size_t)BN_ * ND_);
  float* f_v     = alloc((size_t)BN_ * ND_);
  float* f_big   = alloc((size_t)E_ * 768);      // e0|e1|hsum ; later h_input (E x 608)
  float* f_e0    = f_big;
  float* f_e1    = f_big + (size_t)E_ * ND_;
  float* f_hsum  = f_big + (size_t)E_ * 2 * ND_;
  float* f_hattn = alloc((size_t)BN_ * ND_);
  float* f_hedgemsa = alloc((size_t)E_ * ED_);
  float* f_hnode2   = alloc((size_t)BN_ * ND_);
  float* f_ff1   = alloc((size_t)BN_ * 2 * ND_);
  float* f_ff2   = alloc((size_t)BN_ * ND_);
  float* f_he    = alloc((size_t)E_ * ED_);
  float* f_ffe1  = alloc((size_t)E_ * 2 * ED_);
  float* f_ffe2  = alloc((size_t)E_ * ED_);
  float* f_uc1   = alloc((size_t)E_ * ND_);
  float* f_alpha  = f_cat;                       // E*8   <= E*96, cat dead by then
  float* f_hinput = f_big;                       // E*608 <= E*768, e0/e1/hsum dead
  float* f_ui     = f_et;                        // E*256 <= E*384, et dead by then
  (void)ws_size; (void)in_sizes; (void)n_in; (void)out_size;

  // 1) geometry + padded concat([dist, edge_attr])
  k_edge_geom<<<(E_ + 255) / 256, 256, 0, stream>>>(pos, ei, edge_attr, coors_scale,
                                                    f_cat, f_dist, f_cdiff);
  // 2) ea0 = cat @ W_ee + b_ee                         [E,65]@[65,64]
  GEMM(0, 0, f_cat, W_ee, b_ee, f_ea, E_, ED_, 65, CAT_LD);
  // 3) nt = silu(node_time) @ W_nt + b_nt              [1536,1024]@[1024,1536]
  GEMM(1, 0, node_time, W_nt, b_nt, f_nt, BN_, 6 * ND_, TD_, TD_);
  // 4) et = silu(edge_time) @ W_et + b_et              [E,1024]@[1024,384]
  GEMM(1, 0, edge_time, W_et, b_et, f_et, E_, 6 * ED_, TD_, TD_);
  // 5) ut = silu(edge_time) @ W_ut + b_ut              [E,1024]@[1024,512]
  GEMM(1, 0, edge_time, W_ut, b_ut, f_ut, E_, 2 * ND_, TD_, TD_);
  // 6) hh = modulate(ln(h), n_sh_msa, n_sc_msa)
  k_ln_mod<<<BN_, ND_, ND_ * sizeof(float), stream>>>(h, f_nt, 6 * ND_, 0, ND_,
                                                      nullptr, f_hh, ND_);
  // 7) ea = modulate(ln(ea0), e_sh_msa, e_sc_msa)   (in-place)
  k_ln_mod<<<E_, ED_, ED_ * sizeof(float), stream>>>(f_ea, f_et, 6 * ED_, 0, ED_,
                                                     nullptr, f_ea, ED_);
  // 8) q,k,v and edge projections
  GEMM(0, 0, f_hh, Wq, bq, f_q, BN_, ND_, ND_, ND_);
  GEMM(0, 0, f_hh, Wk, bk, f_k, BN_, ND_, ND_, ND_);
  GEMM(0, 0, f_hh, Wv, bv, f_v, BN_, ND_, ND_, ND_);
  GEMM(0, 0, f_ea, We0, nullptr, f_e0, E_, ND_, ED_, ED_);
  GEMM(0, 0, f_ea, We1, nullptr, f_e1, E_, ND_, ED_, ED_);
  // 9) edge attention logits
  k_alpha<<<(E_ * HEADS_ + 255) / 256, 256, 0, stream>>>(f_q, f_k, f_e0, ei, f_alpha);
  // 10) per-node softmax + aggregation
  k_attn_node<<<BN_, ND_, 0, stream>>>(f_alpha, f_v, f_e1, f_hattn);
  // 11) hsum = h_attn[row] + h_attn[col]
  k_pair_sum<<<((size_t)E_ * ND_ + 255) / 256, 256, 0, stream>>>(f_hattn, ei, f_hsum);
  // 12) h_edge_msa = hsum @ W_n2e + b_n2e              [E,256]@[256,64]
  GEMM(0, 0, f_hsum, W_n2e, b_n2e, f_hedgemsa, E_, ED_, ND_, ND_);
  // 13) h_node2 = modulate(ln(h + g_msa*attn)) * mask
  k_gate_ln_mod<<<BN_, ND_, ND_ * sizeof(float), stream>>>(
      h, f_hattn, f_nt, 6 * ND_, 2 * ND_, 3 * ND_, 4 * ND_, node_mask, f_hnode2, ND_);
  // 14) node FFN
  GEMM(0, 1, f_hnode2, W_ff1, b_ff1, f_ff1, BN_, 2 * ND_, ND_, ND_);
  GEMM(0, 0, f_ff1, W_ff2, b_ff2, f_ff2, BN_, ND_, 2 * ND_, 2 * ND_);
  // 15) h_out
  k_node_out<<<(BN_ * ND_ + 255) / 256, 256, 0, stream>>>(f_hnode2, f_ff2, f_nt,
                                                          node_mask, out_h);
  // 16) h_e = modulate(ln(edge_attr + e_g_msa*h_edge_msa))
  k_gate_ln_mod<<<E_, ED_, ED_ * sizeof(float), stream>>>(
      edge_attr, f_hedgemsa, f_et, 6 * ED_, 2 * ED_, 3 * ED_, 4 * ED_, nullptr, f_he, ED_);
  // 17) edge FFN
  GEMM(0, 1, f_he, W_ff3, b_ff3, f_ffe1, E_, 2 * ED_, ED_, ED_);
  GEMM(0, 0, f_ffe1, W_ff4, b_ff4, f_ffe2, E_, ED_, 2 * ED_, 2 * ED_);
  // 18) h_edge_out
  k_edge_out<<<((size_t)E_ * ED_ + 255) / 256, 256, 0, stream>>>(f_he, f_ffe2, f_et,
                                                                 out_edge);
  // 19) h_input = [h_out[row], h_out[col], h_edge_out, dist] (stride 608, zero pad)
  k_concat_hinput<<<((size_t)E_ * HIN_LD + 255) / 256, 256, 0, stream>>>(
      out_h, out_edge, f_dist, ei, f_hinput);
  // 20) ui = h_input @ W_ui + b_ui                     [E,577]@[577,256]
  GEMM(0, 0, f_hinput, W_ui, b_ui, f_ui, E_, ND_, 577, HIN_LD);
  // 21) inv1 = modulate(ln(ui), t_shift, t_scale)   (in-place)
  k_ln_mod<<<E_, ND_, ND_ * sizeof(float), stream>>>(f_ui, f_ut, 2 * ND_, 0, ND_,
                                                     nullptr, f_ui, ND_);
  // 22) uc1 = silu(inv1 @ W_uc1 + b_uc1)               [E,256]@[256,256]
  GEMM(0, 1, f_ui, W_uc1, b_uc1, f_uc1, E_, ND_, ND_, ND_);
  // 23) pos_out = pos + scatter(cdiff * tanh(uc1 @ W_uc2), row)
  k_copy<<<(BN_ * 3 + 255) / 256, 256, 0, stream>>>(pos, out_pos, BN_ * 3);
  k_inv_scatter<<<(E_ + 255) / 256, 256, 0, stream>>>(f_uc1, W_uc2, f_cdiff, ei, out_pos);
}
